// ConstantDelayGwAC_28123445854586
// MI455X (gfx1250) — compile-verified, matrix operand order checked by reading, not run on hardware
//
#include <hip/hip_runtime.h>
#include <hip/hip_bf16.h>
#include <math.h>

// Problem constants (from reference)
#define NNODES 256
#define DEG    4
#define IN_F   32
#define HID    64
#define MSG    32
#define OUTC   8
#define TSTEPS (5 * NNODES)        // 1280 sequential steps
#define KDIM   (HID + MSG)         // 96
#define CPW    16                  // chains per workgroup (one WMMA row tile)
#define NWG    (NNODES / CPW)      // 16 workgroups

typedef __attribute__((ext_vector_type(16))) _Float16 v16h;
typedef __attribute__((ext_vector_type(8)))  float    v8f;

// ---------------- workspace layout (all f32) ----------------
// encoded : [256][64]                     (64 KB)
// preds   : [256 chains][256 nodes][64]   (16 MB)
// Mhist   : [T][256 chains][32]           (40 MB)
#define WS_ENC_OFF   0
#define WS_PREDS_OFF (NNODES * HID)                        // in floats
#define WS_MHIST_OFF (WS_PREDS_OFF + NNODES * NNODES * HID)

// ---------------- kernel 1: encoded = x @ W_enc + b_enc ----------------
__global__ void cd_encode(const float* __restrict__ x, const float* __restrict__ We,
                          const float* __restrict__ be, float* __restrict__ enc) {
    __shared__ float sx[IN_F];
    const int v = blockIdx.x, h = threadIdx.x;   // 256 blocks x 64 threads
    if (h < IN_F) sx[h] = x[v * IN_F + h];
    __syncthreads();
    float acc = be[h];
#pragma unroll
    for (int i = 0; i < IN_F; ++i) acc += sx[i] * We[i * HID + h];
    enc[v * HID + h] = acc;
}

// ---------------- kernel 2: preds[s][v][:] = encoded[v][:] for all s ----------------
__global__ void cd_bcast(const float* __restrict__ enc, float* __restrict__ preds) {
    const int total = (NNODES * NNODES * HID) / 4;   // float4 count
    const int srcQ  = (NNODES * HID) / 4;            // float4 per encoded copy
    for (int q = blockIdx.x * blockDim.x + threadIdx.x; q < total;
         q += gridDim.x * blockDim.x) {
        ((float4*)preds)[q] = ((const float4*)enc)[q & (srcQ - 1)];
    }
}

// ---------------- main persistent kernel: 16 WGs x 16 chains, 1280 steps ----------------
__global__ __launch_bounds__(256) void cd_bfs_main(
    const int*   __restrict__ neighbors,   // [256][4]
    const float* __restrict__ first_msg,   // [32]
    const float* __restrict__ Wns,         // [96][64]
    const float* __restrict__ bns,         // [64]
    const float* __restrict__ Wnm,         // [96][32]
    const float* __restrict__ bnm,         // [32]
    float* __restrict__ preds,             // [256][256][64]
    float* __restrict__ Mhist)             // [T][256][32]
{
    __shared__ int      sNeigh[NNODES * DEG];          // 4 KB
    __shared__ _Float16 sWnsT[HID * KDIM];             // W_ns^T [n][k]  12 KB (prologue only)
    __shared__ _Float16 sWnmT[MSG * KDIM];             // W_nm^T [n][k]   6 KB (prologue only)
    __shared__ _Float16 sX[CPW * KDIM];                // A matrix [16][96] (feat|msg)
    __shared__ float    sHf[CPW * HID];                // h (f32) for scatter
    __shared__ float    sM[CPW * MSG];                 // m (f32) for Mhist
    __shared__ int      sNode[CPW];
    __shared__ float    sBns[HID], sBnm[MSG], sFm[MSG];

    const int tid    = threadIdx.x;
    const int lane   = tid & 31;
    const int wave   = tid >> 5;            // 8 waves (wave32)
    const int chain0 = blockIdx.x * CPW;

    // ---- preload weights (transposed, f16), biases, graph ----
    for (int i = tid; i < HID * KDIM; i += 256) {
        int n = i / KDIM, k = i % KDIM;
        sWnsT[i] = (_Float16)Wns[k * HID + n];
    }
    for (int i = tid; i < MSG * KDIM; i += 256) {
        int n = i / KDIM, k = i % KDIM;
        sWnmT[i] = (_Float16)Wnm[k * MSG + n];
    }
    for (int i = tid; i < NNODES * DEG; i += 256) sNeigh[i] = neighbors[i];
    if (tid < HID) sBns[tid] = bns[tid];
    if (tid < MSG) { sBnm[tid] = bnm[tid]; sFm[tid] = first_msg[tid]; }
    __syncthreads();

    const int mrow = lane & 15;             // WMMA A row / D col selector
    const int kb   = (lane >> 4) << 3;      // K sub-block base within slice
    const int ncol = (lane & 15) + ((wave & 3) << 4);   // output column (per role)

    // ---- hoist loop-invariant B fragments + biases into registers ----
    // waves 0..3: GEMM1 (W_ns^T), 3 k-slices.  waves 4..5: GEMM2 (W_nm^T), 3 k-slices.
    v16h bW[3];
    float biasReg = 0.0f;
    if (wave < 4) {
        biasReg = sBns[ncol];
#pragma unroll
        for (int kk = 0; kk < 3; ++kk) {
            const _Float16* pb = &sWnsT[ncol * KDIM + kk * 32 + kb];
#pragma unroll
            for (int i = 0; i < 8; ++i) { bW[kk][i] = pb[i]; bW[kk][8 + i] = pb[16 + i]; }
        }
    } else if (wave < 6) {
        biasReg = sBnm[ncol];
#pragma unroll
        for (int kk = 0; kk < 3; ++kk) {
            const _Float16* pb = &sWnmT[ncol * KDIM + kk * 32 + kb];
#pragma unroll
            for (int i = 0; i < 8; ++i) { bW[kk][i] = pb[i]; bW[kk][8 + i] = pb[16 + i]; }
        }
    }
    __syncthreads();

    for (int k = 0; k < TSTEPS; ++k) {
        // ---- phase 1: popped node per chain (pure function of graph + k) ----
        if (tid < CPW) {
            int path[8]; int d = 0; int j = k;
            while (j > 0) { path[d++] = j; j = (j - 1) >> 2; }
            int node = chain0 + tid;        // start node of this chain
            for (int i = d - 1; i >= 0; --i)
                node = sNeigh[node * DEG + ((path[i] - 1) & 3)];
            sNode[tid] = node;
        }
        __syncthreads();

        // ---- phase 2: build A = [feat(64) | msg(32)] as f16 ----
        {   // gather feat: 256 threads -> 16 chains x 16 float4
            const int c = tid >> 4, i4 = (tid & 15) << 2;
            const float4 f = *(const float4*)&preds[(((size_t)(chain0 + c)) * NNODES
                                                     + sNode[c]) * HID + i4];
            _Float16* dst = &sX[c * KDIM + i4];
            dst[0] = (_Float16)f.x; dst[1] = (_Float16)f.y;
            dst[2] = (_Float16)f.z; dst[3] = (_Float16)f.w;
        }
        if (tid < 128) {                    // msg: 16 chains x 8 float4
            const int c = tid >> 3, f4 = (tid & 7) << 2;
            float4 mv;
            if (k == 0) mv = *(const float4*)&sFm[f4];
            else {
                const int kp = (k - 1) >> 2;
                mv = *(const float4*)&Mhist[(((size_t)kp) * NNODES + chain0 + c) * MSG + f4];
            }
            _Float16* dst = &sX[c * KDIM + HID + f4];
            dst[0] = (_Float16)mv.x; dst[1] = (_Float16)mv.y;
            dst[2] = (_Float16)mv.z; dst[3] = (_Float16)mv.w;
        }
        if (tid == 0 && k + 1 < TSTEPS) {   // prefetch next step's message block
            const int kp1 = k >> 2;         // ((k+1)-1)/4
            __builtin_prefetch(&Mhist[(((size_t)kp1) * NNODES + chain0) * MSG], 0, 1);
        }
        __syncthreads();

        // ---- phase 3 (overlapped):
        //   waves 0..3 : acc1 = X @ W_ns          (full K = 0..95)
        //   waves 4..5 : acc2 = msg part of GEMM2 (K-slice 2 = cols 64..95, ready now)
        v8f acc1 = {};
        v8f acc2 = {};
        if (wave < 4) {
#pragma unroll
            for (int kk = 0; kk < 3; ++kk) {
                v16h a;
                const _Float16* pa = &sX[mrow * KDIM + kk * 32 + kb];
#pragma unroll
                for (int i = 0; i < 8; ++i) { a[i] = pa[i]; a[8 + i] = pa[16 + i]; }
                acc1 = __builtin_amdgcn_wmma_f32_16x16x32_f16(
                           false, a, false, bW[kk], (short)0, acc1, false, false);
            }
        } else if (wave < 6) {
            v16h a;
            const _Float16* pa = &sX[mrow * KDIM + 64 + kb];
#pragma unroll
            for (int i = 0; i < 8; ++i) { a[i] = pa[i]; a[8 + i] = pa[16 + i]; }
            acc2 = __builtin_amdgcn_wmma_f32_16x16x32_f16(
                       false, a, false, bW[2], (short)0, acc2, false, false);
        }
        __syncthreads();                    // all reads of sX feat columns done

        // ---- phase 3b: bias + relu; write h (f32 for scatter, f16 into A cols 0..63)
        if (wave < 4) {
            const int mb = (lane >> 4) << 3;
#pragma unroll
            for (int r = 0; r < 8; ++r) {
                float v = fmaxf(acc1[r] + biasReg, 0.0f);
                sHf[(mb + r) * HID + ncol]  = v;
                sX [(mb + r) * KDIM + ncol] = (_Float16)v;   // A := [h | msg]
            }
        }
        __syncthreads();

        // ---- phase 4: waves 4..5 finish GEMM2 over the h columns (K-slices 0,1) ----
        if (wave >= 4 && wave < 6) {
#pragma unroll
            for (int kk = 0; kk < 2; ++kk) {
                v16h a;
                const _Float16* pa = &sX[mrow * KDIM + kk * 32 + kb];
#pragma unroll
                for (int i = 0; i < 8; ++i) { a[i] = pa[i]; a[8 + i] = pa[16 + i]; }
                acc2 = __builtin_amdgcn_wmma_f32_16x16x32_f16(
                           false, a, false, bW[kk], (short)0, acc2, false, false);
            }
            const int mb = (lane >> 4) << 3;
#pragma unroll
            for (int r = 0; r < 8; ++r)
                sM[(mb + r) * MSG + ncol] = acc2[r] + biasReg;
        }
        __syncthreads();

        // ---- phase 5: scatter state + append message history ----
        {   const int c = tid >> 4, i4 = (tid & 15) << 2;
            *(float4*)&preds[(((size_t)(chain0 + c)) * NNODES + sNode[c]) * HID + i4] =
                *(const float4*)&sHf[c * HID + i4];
        }
        if (tid < 128) {
            const int c = tid >> 3, f4 = (tid & 7) << 2;
            *(float4*)&Mhist[(((size_t)k) * NNODES + chain0 + c) * MSG + f4] =
                *(const float4*)&sM[c * MSG + f4];
        }
        __syncthreads();
    }
}

// ---------------- kernel 4: out = log_softmax(preds[s][s] @ W_dec + b_dec) ----------------
__global__ void cd_decode(const float* __restrict__ preds, const float* __restrict__ Wd,
                          const float* __restrict__ bd, float* __restrict__ out) {
    const int s = threadIdx.x;              // 1 block x 256 threads
    const float* h = &preds[(((size_t)s) * NNODES + s) * HID];
    float z[OUTC];
#pragma unroll
    for (int o = 0; o < OUTC; ++o) z[o] = bd[o];
    for (int i = 0; i < HID; ++i) {
        const float hv = h[i];
#pragma unroll
        for (int o = 0; o < OUTC; ++o) z[o] += hv * Wd[i * OUTC + o];
    }
    float mx = z[0];
#pragma unroll
    for (int o = 1; o < OUTC; ++o) mx = fmaxf(mx, z[o]);
    float se = 0.f;
#pragma unroll
    for (int o = 0; o < OUTC; ++o) se += expf(z[o] - mx);
    const float lse = mx + logf(se);
#pragma unroll
    for (int o = 0; o < OUTC; ++o) out[s * OUTC + o] = z[o] - lse;
}

extern "C" void kernel_launch(void* const* d_in, const int* in_sizes, int n_in,
                              void* d_out, int out_size, void* d_ws, size_t ws_size,
                              hipStream_t stream) {
    const float* x     = (const float*)d_in[0];
    const int*   neigh = (const int*)  d_in[1];
    const float* fmsg  = (const float*)d_in[2];
    const float* W_enc = (const float*)d_in[3];
    const float* b_enc = (const float*)d_in[4];
    const float* W_ns  = (const float*)d_in[5];
    const float* b_ns  = (const float*)d_in[6];
    const float* W_nm  = (const float*)d_in[7];
    const float* b_nm  = (const float*)d_in[8];
    const float* W_dec = (const float*)d_in[9];
    const float* b_dec = (const float*)d_in[10];
    float* out = (float*)d_out;

    float* ws    = (float*)d_ws;
    float* enc   = ws + WS_ENC_OFF;
    float* preds = ws + WS_PREDS_OFF;
    float* Mhist = ws + WS_MHIST_OFF;

    cd_encode<<<NNODES, HID, 0, stream>>>(x, W_enc, b_enc, enc);
    cd_bcast<<<2048, 256, 0, stream>>>(enc, preds);
    cd_bfs_main<<<NWG, 256, 0, stream>>>(neigh, fmsg, W_ns, b_ns, W_nm, b_nm,
                                         preds, Mhist);
    cd_decode<<<1, NNODES, 0, stream>>>(preds, W_dec, b_dec, out);
}
